// SQDDPGMixer_35270271435554
// MI455X (gfx1250) — matrix operand order repaired and checked by reading, not compiled, not run on hardware
//
#include <hip/hip_runtime.h>
#include <hip/hip_bf16.h>

// ---------------------------------------------------------------------------
// SQDDPG Shapley mixer, fused for gfx1250 (MI455X), bf16 WMMA path.
//   B=1024, S=16, A=8, NA=16, SD=256, EMBED=256, IN=384, M=B*S*A=131072 rows
//   st@W1 contribution is per-b (128 rows share it) -> hoisted into stw1_kernel
//   and injected as accumulator init; main kernel does only the x-part (K=128)
//   of layer 1 plus layer 2.  64 rows/WG, wave tile 32(M)x64(N), B-frag reuse x2.
// ---------------------------------------------------------------------------

typedef __attribute__((ext_vector_type(16))) __bf16 v16bf;
typedef __attribute__((ext_vector_type(8)))  float  v8f;

union Frag {
    v16bf v;
    uint4 q[2];
};

__device__ __forceinline__ unsigned short f2bf(float f) {
    unsigned u = __float_as_uint(f);
    unsigned r = u + 0x7FFFu + ((u >> 16) & 1u);   // round-to-nearest-even
    return (unsigned short)(r >> 16);
}

__device__ __forceinline__ unsigned hash32(unsigned x) {
    x ^= x >> 16; x *= 0x7feb352du;
    x ^= x >> 15; x *= 0x846ca68bu;
    x ^= x >> 16;
    return x;
}

// ---------------------------------------------------------------------------
// 1) per-(b,s) random permutation of 8 agents: pos[i] = rank, inv[pos] = agent
// ---------------------------------------------------------------------------
__global__ void perm_kernel(unsigned char* __restrict__ posg,
                            unsigned char* __restrict__ invg) {
    int e = blockIdx.x * blockDim.x + threadIdx.x;   // 0 .. 16383 (B*S)
    if (e >= 1024 * 16) return;
    unsigned u[8];
    for (int a = 0; a < 8; ++a)
        u[a] = hash32((unsigned)(e * 8 + a) + 0x9E3779B9u * 42u);
    int pos[8];
    for (int i = 0; i < 8; ++i) {
        int rank = 0;
        for (int j = 0; j < 8; ++j)
            rank += (u[j] < u[i]) || (u[j] == u[i] && j < i);
        pos[i] = rank;
    }
    for (int i = 0; i < 8; ++i) {
        posg[e * 8 + i]      = (unsigned char)pos[i];
        invg[e * 8 + pos[i]] = (unsigned char)i;
    }
}

// ---------------------------------------------------------------------------
// 2) pack fp32 weight rows [k0, k0+K) of W ((k0+K) x N row-major) into bf16
//    WMMA-B layout: packed[(ntile*KC + kchunk)*32 + lane][16]
//    lane: h=lane>>4, n=ntile*16+(lane&15); 16 values at k=8h+j (j<8), 16+8h+(j-8) (j>=8).
// ---------------------------------------------------------------------------
__global__ void pack_kernel(const float* __restrict__ W,
                            unsigned short* __restrict__ out,
                            int k0, int K, int N) {
    int idx = blockIdx.x * blockDim.x + threadIdx.x;
    int KC = K >> 5;
    int total = (N >> 4) * KC * 32;
    if (idx >= total) return;
    int lane  = idx & 31;
    int c     = (idx >> 5) % KC;
    int ntile = (idx >> 5) / KC;
    int h = lane >> 4;
    int n = ntile * 16 + (lane & 15);
    unsigned short* dst = out + (size_t)idx * 16;
    for (int j = 0; j < 16; ++j) {
        int k = k0 + c * 32 + (j < 8 ? h * 8 + j : 16 + h * 8 + (j - 8));
        dst[j] = f2bf(W[(size_t)k * N + n]);
    }
}

// ---------------------------------------------------------------------------
// 3) V-head: v[b] = relu(st[b] @ Vw1 + Vb1) @ Vw2 + Vb2   (1024 rows only)
// ---------------------------------------------------------------------------
__global__ __launch_bounds__(256) void v_kernel(const float* __restrict__ states,
                                                const float* __restrict__ Vw1,
                                                const float* __restrict__ Vb1,
                                                const float* __restrict__ Vw2,
                                                const float* __restrict__ Vb2,
                                                float* __restrict__ vout) {
    __shared__ float stS[256];
    __shared__ float red[256];
    int b = blockIdx.x, t = threadIdx.x;
    stS[t] = states[b * 256 + t];
    __syncthreads();
    float acc = Vb1[t];
    for (int k = 0; k < 256; ++k)
        acc += stS[k] * Vw1[k * 256 + t];          // coalesced column walk
    acc = acc > 0.f ? acc : 0.f;
    red[t] = acc * Vw2[t];
    __syncthreads();
    for (int s = 128; s > 0; s >>= 1) {
        if (t < s) red[t] += red[t + s];
        __syncthreads();
    }
    if (t == 0) vout[b] = red[0] + Vb2[0];
}

// ---------------------------------------------------------------------------
// 3b) stW1[b,:] = st[b] @ W1[0:256,:] + b1   (per-b, shared by 128 rows)
// ---------------------------------------------------------------------------
__global__ __launch_bounds__(256) void stw1_kernel(const float* __restrict__ states,
                                                   const float* __restrict__ W1,
                                                   const float* __restrict__ b1,
                                                   float* __restrict__ stW1) {
    __shared__ float stS[256];
    int b = blockIdx.x, t = threadIdx.x;
    stS[t] = states[b * 256 + t];
    __syncthreads();
    float acc = b1[t];
    for (int k = 0; k < 256; ++k)
        acc += stS[k] * W1[k * 256 + t];           // coalesced column walk
    stW1[b * 256 + t] = acc;
}

// ---------------------------------------------------------------------------
// 4) Fused MLP: 64 rows per workgroup (one b, 8 samples x 8 agents).
//    8 waves = 2 (M-groups of 32 rows) x 4 (N-groups of 64 cols).
//    Each wave: 2 M-tiles x 4 N-tiles -> every B fragment feeds 2 WMMAs.
//    Layer-1 accumulators are seeded with stW1[b][n] (st part precomputed).
// ---------------------------------------------------------------------------
__global__ __launch_bounds__(256) void mlp_kernel(
    const float* __restrict__ stW1, const float* __restrict__ agent_qs,
    const float* __restrict__ b2,
    const float* __restrict__ W3, const float* __restrict__ b3,
    const unsigned short* __restrict__ pW1x, const unsigned short* __restrict__ pW2,
    const unsigned char* __restrict__ posg, const unsigned char* __restrict__ invg,
    float* __restrict__ adv) {

    __shared__ __align__(16) unsigned short xS[64 * 136];    // per-row masked qs
    __shared__ __align__(16) unsigned short hS[64 * 264];    // layer-1 activations
    __shared__ float sw1S[256], b2S[256], w3S[256];
    __shared__ float advPart[4][64];

    const int tid = threadIdx.x;
    const int r0  = blockIdx.x * 64;          // global row base
    const int b   = r0 >> 7;                  // batch row (128 rows per b)
    const int s0  = (r0 >> 3) & 15;           // first sample in this block (0 or 8)

    // ---- stage stW1 row / biases / W3 / masked x -------------------------
    sw1S[tid] = stW1[b * 256 + tid];
    b2S[tid]  = b2[tid];
    w3S[tid]  = W3[tid];
    for (int rr = 0; rr < 2; ++rr) {
        int idx = tid + rr * 256;             // 0..511
        int rl  = idx >> 3;                   // local row 0..63
        int p   = idx & 7;                    // position slot
        int s   = s0 + (rl >> 3);
        int i   = rl & 7;                     // agent index for this row
        int e   = (b * 16 + s) * 8;
        int agent = invg[e + p];
        int keep  = (p <= (int)posg[e + i]) ? 1 : 0;
        const float* q = agent_qs + ((size_t)b * 128 + agent * 16);
        unsigned short* dst = &xS[rl * 136 + p * 16];
        for (int na = 0; na < 16; ++na)
            dst[na] = keep ? f2bf(q[na]) : (unsigned short)0;
    }
    __syncthreads();

    const int lane = tid & 31;
    const int wave = tid >> 5;
    const int h    = lane >> 4;
    const int nl   = lane & 15;
    const int mgrp = wave >> 2;               // 0..1 -> rows mgrp*32 ..
    const int ngrp = wave & 3;                // 0..3 -> cols ngrp*64 ..
    const int row0 = mgrp * 32 + nl;          // A row for M-tile 0
    const int row1 = row0 + 16;               // A row for M-tile 1

    // ---- layer 1 (x part only, K=128): acc seeded with stW1[b][ncol] -----
    v8f acc[2][4];
    for (int t = 0; t < 4; ++t) {
        float seed = sw1S[ngrp * 64 + t * 16 + nl];   // per-column, all rows equal
        for (int r = 0; r < 8; ++r) {
            acc[0][t][r] = seed;
            acc[1][t][r] = seed;
        }
    }
    for (int c = 0; c < 4; ++c) {
        const char* base0 = (const char*)&xS[row0 * 136] + c * 64;
        const char* base1 = (const char*)&xS[row1 * 136] + c * 64;
        Frag a0, a1;
        a0.q[0] = *(const uint4*)(base0 + 16 * h);
        a0.q[1] = *(const uint4*)(base0 + 32 + 16 * h);
        a1.q[0] = *(const uint4*)(base1 + 16 * h);
        a1.q[1] = *(const uint4*)(base1 + 32 + 16 * h);
        for (int t = 0; t < 4; ++t) {
            int ntile = ngrp * 4 + t;
            const uint4* pb = (const uint4*)(pW1x + ((size_t)(ntile * 4 + c) * 32 + lane) * 16);
            Frag bf; bf.q[0] = pb[0]; bf.q[1] = pb[1];
            acc[0][t] = __builtin_amdgcn_wmma_f32_16x16x32_bf16(
                false, a0.v, false, bf.v, (short)0, acc[0][t], false, false);
            acc[1][t] = __builtin_amdgcn_wmma_f32_16x16x32_bf16(
                false, a1.v, false, bf.v, (short)0, acc[1][t], false, false);
        }
    }

    // ---- relu -> bf16 h in LDS (bias already folded into seed) -----------
    for (int mm = 0; mm < 2; ++mm) {
        for (int t = 0; t < 4; ++t) {
            int ncol = ngrp * 64 + t * 16 + nl;
            for (int r = 0; r < 8; ++r) {
                int m = mgrp * 32 + mm * 16 + h * 8 + r;
                float v = acc[mm][t][r];
                v = v > 0.f ? v : 0.f;
                hS[m * 264 + ncol] = f2bf(v);
            }
        }
    }
    __syncthreads();

    // ---- layer 2: K=256 from hS ------------------------------------------
    v8f acc2[2][4] = {};
    for (int c = 0; c < 8; ++c) {
        const char* base0 = (const char*)&hS[row0 * 264] + c * 64;
        const char* base1 = (const char*)&hS[row1 * 264] + c * 64;
        Frag a0, a1;
        a0.q[0] = *(const uint4*)(base0 + 16 * h);
        a0.q[1] = *(const uint4*)(base0 + 32 + 16 * h);
        a1.q[0] = *(const uint4*)(base1 + 16 * h);
        a1.q[1] = *(const uint4*)(base1 + 32 + 16 * h);
        for (int t = 0; t < 4; ++t) {
            int ntile = ngrp * 4 + t;
            const uint4* pb = (const uint4*)(pW2 + ((size_t)(ntile * 8 + c) * 32 + lane) * 16);
            Frag bf; bf.q[0] = pb[0]; bf.q[1] = pb[1];
            acc2[0][t] = __builtin_amdgcn_wmma_f32_16x16x32_bf16(
                false, a0.v, false, bf.v, (short)0, acc2[0][t], false, false);
            acc2[1][t] = __builtin_amdgcn_wmma_f32_16x16x32_bf16(
                false, a1.v, false, bf.v, (short)0, acc2[1][t], false, false);
        }
    }

    // ---- layer 3: relu(h2) . W3, lane-shuffle reduction ------------------
    float p[2][8] = {};
    for (int mm = 0; mm < 2; ++mm) {
        for (int t = 0; t < 4; ++t) {
            int ncol = ngrp * 64 + t * 16 + nl;
            float bias = b2S[ncol];
            float w3v  = w3S[ncol];
            for (int r = 0; r < 8; ++r) {
                float v = acc2[mm][t][r] + bias;
                v = v > 0.f ? v : 0.f;
                p[mm][r] += v * w3v;
            }
        }
    }
    for (int m = 8; m >= 1; m >>= 1)
        for (int mm = 0; mm < 2; ++mm)
            for (int r = 0; r < 8; ++r)
                p[mm][r] += __shfl_xor(p[mm][r], m, 32);  // stays in 16-lane halves

    if (nl == 0) {
        for (int mm = 0; mm < 2; ++mm)
            for (int r = 0; r < 8; ++r)
                advPart[ngrp][mgrp * 32 + mm * 16 + h * 8 + r] = p[mm][r];
    }
    __syncthreads();
    if (tid < 64) {
        float sum = advPart[0][tid] + advPart[1][tid] +
                    advPart[2][tid] + advPart[3][tid] + b3[0];
        adv[r0 + tid] = sum;
    }
}

// ---------------------------------------------------------------------------
// 5) out[b,i] = v[b] + mean_s adv[b,s,i]
// ---------------------------------------------------------------------------
__global__ void finalize_kernel(const float* __restrict__ adv,
                                const float* __restrict__ vv,
                                float* __restrict__ out) {
    int o = blockIdx.x * blockDim.x + threadIdx.x;   // 0..8191
    if (o >= 8192) return;
    int b = o >> 3, i = o & 7;
    float s = 0.f;
    for (int ss = 0; ss < 16; ++ss)
        s += adv[(b * 16 + ss) * 8 + i];
    out[o] = vv[b] + s * (1.f / 16.f);
}

// ---------------------------------------------------------------------------
extern "C" void kernel_launch(void* const* d_in, const int* in_sizes, int n_in,
                              void* d_out, int out_size, void* d_ws, size_t ws_size,
                              hipStream_t stream) {
    (void)in_sizes; (void)n_in; (void)out_size; (void)ws_size;
    const float* states = (const float*)d_in[0];
    const float* aqs    = (const float*)d_in[1];
    const float* W1     = (const float*)d_in[2];
    const float* b1     = (const float*)d_in[3];
    const float* W2     = (const float*)d_in[4];
    const float* b2     = (const float*)d_in[5];
    const float* W3     = (const float*)d_in[6];
    const float* b3     = (const float*)d_in[7];
    const float* Vw1    = (const float*)d_in[8];
    const float* Vb1    = (const float*)d_in[9];
    const float* Vw2    = (const float*)d_in[10];
    const float* Vb2    = (const float*)d_in[11];
    float* out = (float*)d_out;
    char*  ws  = (char*)d_ws;

    // workspace layout (256B aligned)
    constexpr size_t OFF_PW1X = 0;                        // 16*4*32*16*2 = 65536
    constexpr size_t OFF_PW2  = 65536;                    // 16*8*32*16*2 = 131072
    constexpr size_t OFF_ADV  = OFF_PW2 + 131072;         // 131072 * 4   = 524288
    constexpr size_t OFF_V    = OFF_ADV + 524288;         // 1024 * 4
    constexpr size_t OFF_POS  = OFF_V + 4096;             // 16384 * 8
    constexpr size_t OFF_INV  = OFF_POS + 131072;         // 16384 * 8
    constexpr size_t OFF_STW1 = OFF_INV + 131072;         // 1024*256*4 = 1048576

    unsigned short* pW1x = (unsigned short*)(ws + OFF_PW1X);
    unsigned short* pW2  = (unsigned short*)(ws + OFF_PW2);
    float* adv           = (float*)(ws + OFF_ADV);
    float* vbuf          = (float*)(ws + OFF_V);
    unsigned char* posg  = (unsigned char*)(ws + OFF_POS);
    unsigned char* invg  = (unsigned char*)(ws + OFF_INV);
    float* stW1          = (float*)(ws + OFF_STW1);

    perm_kernel<<<64, 256, 0, stream>>>(posg, invg);
    pack_kernel<<<8, 256, 0, stream>>>(W1, pW1x, 256, 128, 256);  // x rows of W1
    pack_kernel<<<16, 256, 0, stream>>>(W2, pW2, 0, 256, 256);
    v_kernel<<<1024, 256, 0, stream>>>(states, Vw1, Vb1, Vw2, Vb2, vbuf);
    stw1_kernel<<<1024, 256, 0, stream>>>(states, W1, b1, stW1);
    mlp_kernel<<<2048, 256, 0, stream>>>(stW1, aqs, b2, W3, b3,
                                         pW1x, pW2, posg, invg, adv);
    finalize_kernel<<<32, 256, 0, stream>>>(adv, vbuf, out);
}
